// GNNDecoder_13486197310273
// MI455X (gfx1250) — matrix operand order repaired and checked by reading, not compiled, not run on hardware
//
#include <hip/hip_runtime.h>
#include <math.h>

// ---------- types ----------
typedef __attribute__((ext_vector_type(16))) __bf16        v16bf;
typedef __attribute__((ext_vector_type(8)))  float         v8f;
typedef __attribute__((ext_vector_type(4)))  unsigned int  uint4v;

union Frag { uint4v q[2]; v16bf v; };

__device__ __forceinline__ unsigned short f2bf(float f) {
    union { float f; unsigned int u; } v; v.f = f;
    unsigned int u = v.u;
    u += 0x7FFFu + ((u >> 16) & 1u);   // round-to-nearest-even
    return (unsigned short)(u >> 16);
}

// ---------- elementwise kernels ----------
__global__ void cvt_bf16_kernel(const float* __restrict__ s, unsigned short* __restrict__ d, int n) {
    int i = blockIdx.x * 256 + threadIdx.x;
    if (i < n) d[i] = f2bf(s[i]);
}

__global__ void zero_f32_kernel(float* p, int n) {
    int i = blockIdx.x * 256 + threadIdx.x;
    if (i < n) p[i] = 0.0f;
}

__global__ void zero_u16_kernel(unsigned short* p, int n) {
    int i = blockIdx.x * 256 + threadIdx.x;
    if (i < n) p[i] = 0;
}

// X0[e, 0:128] = h_bf[src[e]]; X0[e, 128:256] = h_bf[dst[e]]  (16B chunks)
__global__ void gather_kernel(const unsigned short* __restrict__ hbf,
                              const int* __restrict__ src, const int* __restrict__ dst,
                              unsigned short* __restrict__ x0) {
    int idx = blockIdx.x * 256 + threadIdx.x;   // E*32 threads (8 elems each)
    int e = idx >> 5, c = idx & 31;
    int node = (c < 16) ? src[e] : dst[e];
    int jc = c & 15;
    *(uint4v*)(x0 + (size_t)e * 256 + c * 8) =
        *(const uint4v*)(hbf + (size_t)node * 128 + jc * 8);
}

// agg[dst[e], j] += msg[e, j]
__global__ void scatter_kernel(const float* __restrict__ msg, const int* __restrict__ dst,
                               float* __restrict__ agg) {
    int idx = blockIdx.x * 256 + threadIdx.x;   // E*128 threads
    int e = idx >> 7, j = idx & 127;
    unsafeAtomicAdd(&agg[(size_t)dst[e] * 128 + j], msg[(size_t)e * 128 + j]);
}

// xg (N,192) bf16 = concat(agg (N,128) f32, node_in (N,64) f32)
__global__ void concat_kernel(const float* __restrict__ agg, const float* __restrict__ node_in,
                              unsigned short* __restrict__ xg) {
    int idx = blockIdx.x * 256 + threadIdx.x;   // N*192 threads
    int i = idx / 192, j = idx - i * 192;
    float v = (j < 128) ? agg[(size_t)i * 128 + j] : node_in[(size_t)i * 64 + (j - 128)];
    xg[idx] = f2bf(v);
}

// GRU gates (torch order r,z,n), updates h (f32) and hbf (bf16)
__global__ void gru_kernel(const float* __restrict__ gi, const float* __restrict__ gh,
                           float* __restrict__ h, unsigned short* __restrict__ hbf) {
    int idx = blockIdx.x * 256 + threadIdx.x;   // N*128 threads
    int i = idx >> 7, c = idx & 127;
    size_t b = (size_t)i * 384 + c;
    float ir = gi[b], iz = gi[b + 128], in_ = gi[b + 256];
    float hr = gh[b], hz = gh[b + 128], hn  = gh[b + 256];
    float r = 1.0f / (1.0f + __expf(-(ir + hr)));
    float z = 1.0f / (1.0f + __expf(-(iz + hz)));
    float n = tanhf(in_ + r * hn);
    float hv = (1.0f - z) * n + z * h[idx];
    h[idx]   = hv;
    hbf[idx] = f2bf(hv);
}

// ---------- WMMA GEMM: out(M,N) = act( X(M,K)bf16 @ W(N,K)^T bf16 + bias ) ----------
// Block = 256 threads = 8 waves. Block tile: M=128 x N=128.
// W tile (<=128 rows x K) staged in LDS (rows padded by PAD elems -> conflict-free
// ds_load_b128). Wave (wm = w&3, wn = w>>2) computes a 32(M) x 64(N) register tile:
// 2 M-subtiles x 4 N-tiles = 8 wmma accumulators; one B fragment feeds 2 WMMAs.
// A is double-buffered from global (full K-step lead); B uses a rolling 2-fragment
// buffer from LDS (one fragment = 2 ds_load_b128, issued 2 WMMAs ahead).
// K loop fully unrolled (template). Requires M%128==0, N%64==0, K%32==0.
template <int RELU, int OUTBF, int K>
__global__ void gemm_bf16_wmma(const unsigned short* __restrict__ X,
                               const unsigned short* __restrict__ W,
                               const float* __restrict__ bias,
                               void* __restrict__ out,
                               int M, int N) {
    constexpr int PAD    = 8;          // elements of row padding in LDS
    constexpr int LROW   = K + PAD;    // LDS row stride (elements)
    constexpr int NSTEP  = K / 32;
    extern __shared__ unsigned short sm[];

    const int tid  = threadIdx.x;
    const int lane = tid & 31;
    const int w    = tid >> 5;
    const int l15  = lane & 15;
    const int half = lane >> 4;

    const int n0_block = blockIdx.x * 128;
    const int m0       = blockIdx.y * 128 + (w & 3) * 32;
    const int n_base   = n0_block + (w >> 2) * 64;

    // ---- stage W tile into LDS (16B chunks) ----
    {
        const int nt     = min(128, N - n0_block);   // valid rows in this tile
        const int chunks = nt * (K / 8);
        for (int c = tid; c < chunks; c += 256) {
            const int row = c / (K / 8);
            const int cc  = c - row * (K / 8);
            *(uint4v*)(sm + (size_t)row * LROW + cc * 8) =
                *(const uint4v*)(W + (size_t)(n0_block + row) * K + cc * 8);
        }
    }
    __syncthreads();

    if (n_base + 64 > N) return;   // wave-uniform (only hits when N==64)

    v8f acc[2][4];
#pragma unroll
    for (int s = 0; s < 2; ++s)
#pragma unroll
        for (int j = 0; j < 4; ++j) acc[s][j] = (v8f){0.f,0.f,0.f,0.f,0.f,0.f,0.f,0.f};

    // A base pointers for the two 16-row subtiles
    const unsigned short* arow0 = X + (size_t)(m0 + l15)      * K + (half << 3);
    const unsigned short* arow1 = X + (size_t)(m0 + 16 + l15) * K + (half << 3);
    // LDS base index for this wave's B columns (n-tile j adds j*16*LROW, k-step adds k)
    const int bcol = ((w >> 2) * 64 + l15) * LROW + (half << 3);

    Frag a[2][2];   // [buffer][subtile], double-buffered over K-steps
    Frag bb[2];     // rolling 2-fragment B buffer

    // preload A k-step 0
    a[0][0].q[0] = *(const uint4v*)(arow0);
    a[0][0].q[1] = *(const uint4v*)(arow0 + 16);
    a[0][1].q[0] = *(const uint4v*)(arow1);
    a[0][1].q[1] = *(const uint4v*)(arow1 + 16);
    // preload B fragment 0 (s=0, j=0)
    bb[0].q[0] = *(const uint4v*)(sm + bcol);
    bb[0].q[1] = *(const uint4v*)(sm + bcol + 16);

#pragma unroll
    for (int s = 0; s < NSTEP; ++s) {
        const int curA = s & 1, nxtA = curA ^ 1;
        if (s + 1 < NSTEP) {                       // prefetch next A fragments
            const int kn = (s + 1) * 32;
            a[nxtA][0].q[0] = *(const uint4v*)(arow0 + kn);
            a[nxtA][0].q[1] = *(const uint4v*)(arow0 + kn + 16);
            a[nxtA][1].q[0] = *(const uint4v*)(arow1 + kn);
            a[nxtA][1].q[1] = *(const uint4v*)(arow1 + kn + 16);
        }
#pragma unroll
        for (int j = 0; j < 4; ++j) {
            const int fi  = s * 4 + j;            // linear fragment index
            const int cur = fi & 1, nxt = cur ^ 1;
            if (fi + 1 < NSTEP * 4) {             // prefetch next B fragment
                const int fj = fi + 1;
                const int ss = fj >> 2, jj = fj & 3;
                const unsigned short* bp = sm + bcol + jj * 16 * LROW + ss * 32;
                bb[nxt].q[0] = *(const uint4v*)(bp);
                bb[nxt].q[1] = *(const uint4v*)(bp + 16);
            }
            acc[0][j] = __builtin_amdgcn_wmma_f32_16x16x32_bf16(
                false, a[curA][0].v, false, bb[cur].v, (short)0, acc[0][j], false, false);
            acc[1][j] = __builtin_amdgcn_wmma_f32_16x16x32_bf16(
                false, a[curA][1].v, false, bb[cur].v, (short)0, acc[1][j], false, false);
        }
    }

    // ---- epilogue ----
#pragma unroll
    for (int s = 0; s < 2; ++s) {
        const int mrow = m0 + s * 16 + (half << 3);
#pragma unroll
        for (int j = 0; j < 4; ++j) {
            const int ncol = n_base + j * 16 + l15;
            const float bv = bias[ncol];
#pragma unroll
            for (int r = 0; r < 8; ++r) {
                float v = acc[s][j][r] + bv;
                if (RELU) v = fmaxf(v, 0.0f);
                size_t off = (size_t)(mrow + r) * N + ncol;
                if (OUTBF) ((unsigned short*)out)[off] = f2bf(v);
                else       ((float*)out)[off] = v;
            }
        }
    }
}

// ---------- host ----------
extern "C" void kernel_launch(void* const* d_in, const int* in_sizes, int n_in,
                              void* d_out, int out_size, void* d_ws, size_t ws_size,
                              hipStream_t stream) {
    constexpr int NN = 16384, NEDGE = 65536;
    constexpr int NF = 128, NI = 64, NE = 128, NO = 64, MS = 256, NITERS = 7;

    const float* node_in = (const float*)d_in[0];
    const int*   src     = (const int*)d_in[1];
    const int*   dst     = (const int*)d_in[2];
    const float* W1 = (const float*)d_in[3];   const float* b1 = (const float*)d_in[4];
    const float* W2 = (const float*)d_in[5];   const float* b2 = (const float*)d_in[6];
    const float* W3 = (const float*)d_in[7];   const float* b3 = (const float*)d_in[8];
    const float* W4 = (const float*)d_in[9];   const float* b4 = (const float*)d_in[10];
    const float* Wih = (const float*)d_in[11]; const float* Whh = (const float*)d_in[12];
    const float* bih = (const float*)d_in[13]; const float* bhh = (const float*)d_in[14];
    const float* Wf  = (const float*)d_in[15]; const float* bf  = (const float*)d_in[16];
    float* out = (float*)d_out;

    // workspace carve-up (256B aligned slices)
    char* ws = (char*)d_ws;
    size_t off = 0;
    auto take = [&](size_t bytes) -> char* {
        char* p = ws + off;
        off += (bytes + 255) & ~(size_t)255;
        return p;
    };
    unsigned short* w1b  = (unsigned short*)take((size_t)MS * 2 * NF * 2);
    unsigned short* w2b  = (unsigned short*)take((size_t)MS * MS * 2);
    unsigned short* w3b  = (unsigned short*)take((size_t)MS * MS * 2);
    unsigned short* w4b  = (unsigned short*)take((size_t)NE * MS * 2);
    unsigned short* wihb = (unsigned short*)take((size_t)3 * NF * (NE + NI) * 2);
    unsigned short* whhb = (unsigned short*)take((size_t)3 * NF * NF * 2);
    unsigned short* wfb  = (unsigned short*)take((size_t)NO * NF * 2);
    float*          h    = (float*)take((size_t)NN * NF * 4);
    unsigned short* hbf  = (unsigned short*)take((size_t)NN * NF * 2);
    unsigned short* bufA = (unsigned short*)take((size_t)NEDGE * 256 * 2);  // also MSG f32 (E,128)
    unsigned short* bufB = (unsigned short*)take((size_t)NEDGE * 256 * 2);
    float*          agg  = (float*)take((size_t)NN * NF * 4);
    unsigned short* xg   = (unsigned short*)take((size_t)NN * (NE + NI) * 2);
    float*          gi   = (float*)take((size_t)NN * 3 * NF * 4);
    float*          gh   = (float*)take((size_t)NN * 3 * NF * 4);
    (void)ws_size; (void)n_in; (void)in_sizes; (void)out_size;

    dim3 blk(256);
    auto nb = [](int n) { return (n + 255) / 256; };

    // one-time: weights -> bf16, h0 = 0
    cvt_bf16_kernel<<<nb(MS * 2 * NF), blk, 0, stream>>>(W1, w1b, MS * 2 * NF);
    cvt_bf16_kernel<<<nb(MS * MS), blk, 0, stream>>>(W2, w2b, MS * MS);
    cvt_bf16_kernel<<<nb(MS * MS), blk, 0, stream>>>(W3, w3b, MS * MS);
    cvt_bf16_kernel<<<nb(NE * MS), blk, 0, stream>>>(W4, w4b, NE * MS);
    cvt_bf16_kernel<<<nb(3 * NF * (NE + NI)), blk, 0, stream>>>(Wih, wihb, 3 * NF * (NE + NI));
    cvt_bf16_kernel<<<nb(3 * NF * NF), blk, 0, stream>>>(Whh, whhb, 3 * NF * NF);
    cvt_bf16_kernel<<<nb(NO * NF), blk, 0, stream>>>(Wf, wfb, NO * NF);
    zero_f32_kernel<<<nb(NN * NF), blk, 0, stream>>>(h, NN * NF);
    zero_u16_kernel<<<nb(NN * NF), blk, 0, stream>>>(hbf, NN * NF);

    // GEMM launch helpers: grid = (N/128, M/128), dynamic LDS = min(128,N)*(K+8)*2
    auto ggrid = [](int M, int N) { return dim3((unsigned)((N + 127) / 128), (unsigned)(M / 128)); };
    auto lds   = [](int N, int K) { return (size_t)((N < 128 ? N : 128) * (K + 8) * 2); };

    for (int t = 0; t < NITERS; ++t) {
        // 1) gather edge features (E,256) bf16, 16B chunks
        gather_kernel<<<NEDGE * 32 / 256, blk, 0, stream>>>(hbf, src, dst, bufA);
        // 2-4) msg MLP hidden layers (ReLU, bf16 out), K=256
        gemm_bf16_wmma<1, 1, 256><<<ggrid(NEDGE, MS), blk, lds(MS, 256), stream>>>(bufA, w1b, b1, bufB, NEDGE, MS);
        gemm_bf16_wmma<1, 1, 256><<<ggrid(NEDGE, MS), blk, lds(MS, 256), stream>>>(bufB, w2b, b2, bufA, NEDGE, MS);
        gemm_bf16_wmma<1, 1, 256><<<ggrid(NEDGE, MS), blk, lds(MS, 256), stream>>>(bufA, w3b, b3, bufB, NEDGE, MS);
        // 5) messages (E,128) f32 into bufA
        gemm_bf16_wmma<0, 0, 256><<<ggrid(NEDGE, NE), blk, lds(NE, 256), stream>>>(bufB, w4b, b4, (void*)bufA, NEDGE, NE);
        // 6) segment-sum into agg
        zero_f32_kernel<<<nb(NN * NF), blk, 0, stream>>>(agg, NN * NF);
        scatter_kernel<<<NEDGE * 128 / 256, blk, 0, stream>>>((const float*)bufA, dst, agg);
        // 7) concat [agg, node_inputs] -> bf16 (N,192)
        concat_kernel<<<NN * 192 / 256, blk, 0, stream>>>(agg, node_in, xg);
        // 8-9) GRU gate GEMMs (f32 out)
        gemm_bf16_wmma<0, 0, 192><<<ggrid(NN, 3 * NF), blk, lds(3 * NF, 192), stream>>>(xg, wihb, bih, (void*)gi, NN, 3 * NF);
        gemm_bf16_wmma<0, 0, 128><<<ggrid(NN, 3 * NF), blk, lds(3 * NF, 128), stream>>>(hbf, whhb, bhh, (void*)gh, NN, 3 * NF);
        // 10) GRU elementwise update
        gru_kernel<<<NN * 128 / 256, blk, 0, stream>>>(gi, gh, h, hbf);
        // 11) output projection straight into d_out
        gemm_bf16_wmma<0, 0, 128><<<ggrid(NN, NO), blk, lds(NO, 128), stream>>>(hbf, wfb, bf,
            (void*)(out + (size_t)t * NN * NO), NN, NO);
    }
}
